// DPMPModel_60593398612115
// MI455X (gfx1250) — matrix operand order repaired and checked by reading, not compiled, not run on hardware
//
#include <hip/hip_runtime.h>
#include <math.h>

// ---------------------------------------------------------------------------
// CDNA5 / gfx1250 implementation of the DeePMD-style pair-message energy model.
// All K>=16 matmuls run on v_wmma_f32_16x16x32_f16 (f16 in, f32 accumulate).
// Pair-space intermediates (embed, embMP, embR) stored as f16 to halve HBM
// traffic (model is bandwidth bound: ~300MB moved vs ~7 GFLOP of math).
// Activations use the gfx1250 hardware tanh (single trans-pipe op) instead of
// the branchy libm expansion.
// ---------------------------------------------------------------------------

typedef _Float16 half_t;
typedef half_t v16h __attribute__((ext_vector_type(16)));
typedef float  v8f  __attribute__((ext_vector_type(8)));

#define Nn    768
#define NNp   (768 * 768)
#define HALFN 384

static __device__ __forceinline__ v8f wmma16(v16h a, v16h b, v8f c) {
  // D = A(16x32 f16) * B(32x16 f16) + C(16x16 f32)
  return __builtin_amdgcn_wmma_f32_16x16x32_f16(false, a, false, b, (short)0, c,
                                                false, false);
}
// A-fragment element -> K index (16-bit A 16x32 layout)
static __device__ __forceinline__ int kmapA(int e, int half) {
  return (e & 7) + (half << 3) + ((e >> 3) << 4);
}

// gfx1250 has a hardware tanh transcendental (v_tanh_f32).
static __device__ __forceinline__ float fast_tanh(float x) {
#if __has_builtin(__builtin_amdgcn_tanhf)
  return __builtin_amdgcn_tanhf(x);
#else
  float y;
  // trans-op result hazard: 1 independent op / V_NOP before use
  asm("v_tanh_f32 %0, %1\n\tv_nop" : "=v"(y) : "v"(x));
  return y;
#endif
}
// cos(pi*u) via hardware cosine (input in turns: cos(2*pi*s) = v_cos(s))
static __device__ __forceinline__ float cos_pi(float u) {
#if __has_builtin(__builtin_amdgcn_cosf)
  return __builtin_amdgcn_cosf(0.5f * u);
#else
  return cosf(3.14159265358979f * u);
#endif
}

// ======================= structs for kernel arguments ======================
struct EmbW  { const float *W0, *W1, *b0, *b1; };
struct EmbRW { const float *W0, *W1, *W2, *b0, *b1, *b2; };
struct EmbedArgs {
  EmbW  A[2][2];
  EmbW  Bw[2][2];
  EmbRW R[2][2];
  const float* s_cn;
  half_t* embed;   // [N][N][64] f16
  half_t* embR;    // [N][N][16] f16
};
struct MPW { const float *W0, *W1, *W2, *b0, *b1, *b2, *dt; };
struct MPArgs {
  MPW Mw[2][2];
  const float* RX;       // [10][N][N]
  const half_t* embR;    // [N][N][16]
  const float *A1, *A2;  // [N][2][16]
  const float *B1, *B2;  // [N][2][3][16]
  half_t* out;           // [N][N][64]
};
struct ProjArgs {
  const float* D2;      // [N][2][128]
  const float* T4;      // [N][16][64]
  const float* lin1[2];
  const float* lin3[2];
  const float* n1v[2];
  const float* n3v[2];
  float *A1, *A2, *B1, *B2;
};
struct FitW { const float *W0, *W1, *W2, *b0, *b1, *b2; };
struct FitArgs {
  FitW Fw[2];
  const float* Eb;
  const float* G;       // [N][768]
  float* atomE;         // [N]
};

// ============================ kernel 1: geometry ===========================
__global__ __launch_bounds__(256) void k_geom(const float* __restrict__ coord,
                                              const float* __restrict__ box,
                                              float* __restrict__ s_cn,
                                              float* __restrict__ RX) {
  int t = blockIdx.x * blockDim.x + threadIdx.x;
  if (t >= NNp) return;
  int i = t / Nn, j = t % Nn;
  float L0 = box[0], L1 = box[4], L2 = box[8];
  float d0 = coord[0 * Nn + j] - coord[0 * Nn + i];
  float d1 = coord[1 * Nn + j] - coord[1 * Nn + i];
  float d2 = coord[2 * Nn + j] - coord[2 * Nn + i];
  d0 -= L0 * rintf(d0 / L0);
  d1 -= L1 * rintf(d1 / L1);
  d2 -= L2 * rintf(d2 / L2);
  float r2 = d0 * d0 + d1 * d1 + d2 * d2;
  float r  = (r2 > 4e-30f) ? sqrtf(r2) : 0.0f;
  float u  = fminf(fmaxf(r * (1.0f / 6.0f), 0.0f), 1.0f);
  float sw = 0.5f * cos_pi(u) + 0.5f;
  float sr = ((r > 1e-12f) && (r < 6.0f)) ? (sw / r) : 0.0f;
  int ti = (i >= HALFN) ? 1 : 0;
  const float MEANv[2] = {0.12f, 0.10f};
  const float STDv[2]  = {0.07f, 0.06f};
  s_cn[t] = (sr - MEANv[ti]) / STDv[ti];
  float srn = sr / STDv[ti];
  float inv = (r > 2e-15f) ? (1.0f / (r + 1e-16f)) : 0.0f;
  float x0 = d0 * inv, x1 = d1 * inv, x2 = d2 * inv;
  float m2 = (x0 * x0 + x1 * x1 + x2 * x2) * (1.0f / 3.0f);
  const float S3 = 1.73205080757f, S18 = 4.24264068712f;
  RX[0 * NNp + t] = srn;
  RX[1 * NNp + t] = S3 * srn * x0;
  RX[2 * NNp + t] = S3 * srn * x1;
  RX[3 * NNp + t] = S3 * srn * x2;
  RX[4 * NNp + t] = 3.0f * srn * (x0 * x0 - m2);
  RX[5 * NNp + t] = 3.0f * srn * (x1 * x1 - m2);
  RX[6 * NNp + t] = 3.0f * srn * (x2 * x2 - m2);
  RX[7 * NNp + t] = S18 * srn * x0 * x1;
  RX[8 * NNp + t] = S18 * srn * x1 * x2;
  RX[9 * NNp + t] = S18 * srn * x2 * x0;
}

// ================= kernel 2: embA/embB/embR nets over pairs ================
// one wave per 16-pair tile (same row n); 7 WMMAs per tile
__global__ __launch_bounds__(256) void k_embed(EmbedArgs g) {
  __shared__ float bufA[8][16][32];
  __shared__ float bufB[8][16][32];
  const int w = threadIdx.x >> 5, lane = threadIdx.x & 31;
  const int half = lane >> 4, p = lane & 15, nn = lane & 15;
  const int n = blockIdx.y;
  const int m0 = blockIdx.x * 128 + w * 16;
  const int ti = (n >= HALFN) ? 1 : 0, tj = (m0 >= HALFN) ? 1 : 0;
  const int m = m0 + p;
  const float s = g.s_cn[n * Nn + m];
  float (*BA)[32] = bufA[w];
  float (*BB)[32] = bufB[w];

  // ---------------- embA : 1 -> 16 -> 32 (concat residual) ----------------
  {
    const EmbW wA = g.A[ti][tj];
    v16h a;
#pragma unroll
    for (int k = 0; k < 8; ++k) {
      int c = half * 8 + k;
      float h = fast_tanh(s * wA.W0[c] + wA.b0[c]);
      BA[p][c] = h;
      a[k] = (half_t)h;
    }
#pragma unroll
    for (int k = 8; k < 16; ++k) a[k] = (half_t)0.f;
    __syncthreads();
#pragma unroll
    for (int cb = 0; cb < 2; ++cb) {
      v16h b;
#pragma unroll
      for (int e = 0; e < 16; ++e) {
        int K = kmapA(e, half);
        b[e] = (K < 16) ? (half_t)wA.W1[K * 32 + cb * 16 + nn] : (half_t)0.f;
      }
      v8f acc = {};
      acc = wmma16(a, b, acc);
#pragma unroll
      for (int v = 0; v < 8; ++v) {
        int row = v + 8 * half;
        int ch = cb * 16 + nn;
        float val = BA[row][ch & 15] + fast_tanh(acc[v] + wA.b1[ch]);
        g.embed[((size_t)n * Nn + m0 + row) * 64 + ch] = (half_t)val;
      }
    }
    __syncthreads();
  }
  // ---------------- embB ----------------
  {
    const EmbW wB = g.Bw[ti][tj];
    v16h a;
#pragma unroll
    for (int k = 0; k < 8; ++k) {
      int c = half * 8 + k;
      float h = fast_tanh(s * wB.W0[c] + wB.b0[c]);
      BA[p][c] = h;
      a[k] = (half_t)h;
    }
#pragma unroll
    for (int k = 8; k < 16; ++k) a[k] = (half_t)0.f;
    __syncthreads();
#pragma unroll
    for (int cb = 0; cb < 2; ++cb) {
      v16h b;
#pragma unroll
      for (int e = 0; e < 16; ++e) {
        int K = kmapA(e, half);
        b[e] = (K < 16) ? (half_t)wB.W1[K * 32 + cb * 16 + nn] : (half_t)0.f;
      }
      v8f acc = {};
      acc = wmma16(a, b, acc);
#pragma unroll
      for (int v = 0; v < 8; ++v) {
        int row = v + 8 * half;
        int ch = cb * 16 + nn;
        float val = BA[row][ch & 15] + fast_tanh(acc[v] + wB.b1[ch]);
        g.embed[((size_t)n * Nn + m0 + row) * 64 + 32 + ch] = (half_t)val;
      }
    }
    __syncthreads();
  }
  // -------- embR : 1 -> 16 -> 32 (concat residual) -> 16 (linear) ---------
  {
    const EmbRW wR = g.R[ti][tj];
    v16h a;
#pragma unroll
    for (int k = 0; k < 8; ++k) {
      int c = half * 8 + k;
      float h = fast_tanh(s * wR.W0[c] + wR.b0[c]);
      BA[p][c] = h;
      a[k] = (half_t)h;
    }
#pragma unroll
    for (int k = 8; k < 16; ++k) a[k] = (half_t)0.f;
    __syncthreads();
#pragma unroll
    for (int cb = 0; cb < 2; ++cb) {
      v16h b;
#pragma unroll
      for (int e = 0; e < 16; ++e) {
        int K = kmapA(e, half);
        b[e] = (K < 16) ? (half_t)wR.W1[K * 32 + cb * 16 + nn] : (half_t)0.f;
      }
      v8f acc = {};
      acc = wmma16(a, b, acc);
#pragma unroll
      for (int v = 0; v < 8; ++v) {
        int row = v + 8 * half;
        int ch = cb * 16 + nn;
        BB[row][ch] = BA[row][ch & 15] + fast_tanh(acc[v] + wR.b1[ch]);
      }
    }
    __syncthreads();
    // layer3: 32 -> 16, linear
    v16h a3, b3;
#pragma unroll
    for (int e = 0; e < 16; ++e) a3[e] = (half_t)BB[p][kmapA(e, half)];
#pragma unroll
    for (int e = 0; e < 16; ++e) b3[e] = (half_t)wR.W2[kmapA(e, half) * 16 + nn];
    v8f acc = {};
    acc = wmma16(a3, b3, acc);
#pragma unroll
    for (int v = 0; v < 8; ++v) {
      int row = v + 8 * half;
      float val = acc[v] + wR.b2[nn];
      g.embR[((size_t)n * Nn + m0 + row) * 16 + nn] = (half_t)val;
    }
    __syncthreads();
  }
}

// ======= kernel 3/6: T[n][x][b] = sum_m RX[x][n][m]*emb[n][m][b] / 64 ======
// one wave per atom, 4 column-block f32 accumulators, K=768 in 24 chunks
__global__ __launch_bounds__(256) void k_contract(const float* __restrict__ RX,
                                                  const half_t* __restrict__ emb,
                                                  float* __restrict__ Tout,
                                                  int XR) {
  const int w = threadIdx.x >> 5, lane = threadIdx.x & 31;
  const int half = lane >> 4, p = lane & 15, nn = lane & 15;
  const int n = blockIdx.x * 8 + w;
  v8f acc[4] = {};
  for (int kc = 0; kc < 24; ++kc) {
    // prefetch next K-chunk of the f16 pair tensor (streaming, 23.3 TB/s HBM)
    if (kc + 1 < 24)
      __builtin_prefetch(&emb[((size_t)n * Nn + (kc + 1) * 32 + lane) * 64], 0, 1);
    v16h a;
#pragma unroll
    for (int e = 0; e < 16; ++e) {
      int K = kmapA(e, half);
      int m = kc * 32 + K;
      a[e] = (p < XR) ? (half_t)RX[(size_t)p * NNp + n * Nn + m] : (half_t)0.f;
    }
#pragma unroll
    for (int cb = 0; cb < 4; ++cb) {
      v16h b;
#pragma unroll
      for (int e = 0; e < 16; ++e) {
        int m = kc * 32 + kmapA(e, half);
        b[e] = emb[((size_t)n * Nn + m) * 64 + cb * 16 + nn];
      }
      acc[cb] = wmma16(a, b, acc[cb]);
    }
  }
#pragma unroll
  for (int cb = 0; cb < 4; ++cb)
#pragma unroll
    for (int v = 0; v < 8; ++v) {
      int row = v + 8 * half;
      Tout[(size_t)n * 1024 + row * 64 + cb * 16 + nn] = acc[cb][v] * (1.0f / 64.0f);
    }
}

// ============ kernel 4a: T_2ND[n][t][a*32+c] (per-atom 128-vec) ============
__global__ __launch_bounds__(256) void k_d2(const float* __restrict__ T4,
                                            float* __restrict__ D2) {
  int id = blockIdx.x * blockDim.x + threadIdx.x;
  if (id >= Nn * 256) return;
  int n = id >> 8, rem = id & 255;
  int t = rem >> 7, d = rem & 127;
  int a = d >> 5, c = d & 31;
  const float* Tb = T4 + (size_t)n * 1024;
  float s = Tb[t * 32 + c] * Tb[t * 32 + a];
#pragma unroll
  for (int x = 0; x < 3; ++x)
    s += Tb[(1 + x) * 64 + t * 32 + c] * Tb[(1 + x) * 64 + t * 32 + a];
  D2[(size_t)n * 256 + t * 128 + d] = s;
}

// ===== kernel 4b: per-atom projections A1/A2 (lin1) and B1/B2 (lin3) =======
__global__ __launch_bounds__(256) void k_proj(ProjArgs g) {
  int id = blockIdx.x * blockDim.x + threadIdx.x;
  if (id >= Nn * 2 * 16) return;
  int n = id >> 5, rem = id & 31;
  int j = rem >> 4, e = rem & 15;
  int tn = (n >= HALFN) ? 1 : 0;
  const float* L1 = g.lin1[tn];
  const float* L3 = g.lin3[tn];
  const float* n1 = g.n1v[tn];
  const float* n3 = g.n3v[tn];
  // A1: t=0, k=j  (atom as center)
  {
    float sc = n1[0 * 2 + j]; sc *= sc;
    const float* D = g.D2 + (size_t)n * 256;
    const float* W = L1 + ((0 * 2 + j) * 128) * 16 + e;
    float s = 0.f;
    for (int d = 0; d < 128; ++d) s += D[d] * W[d * 16];
    g.A1[(n * 2 + j) * 16 + e] = s * sc;
  }
  // A2: t=1, k=j  (atom as neighbor)
  {
    float sc = n1[1 * 2 + j]; sc *= sc;
    const float* D = g.D2 + (size_t)n * 256 + 128;
    const float* W = L1 + ((1 * 2 + j) * 128) * 16 + e;
    float s = 0.f;
    for (int d = 0; d < 128; ++d) s += D[d] * W[d * 16];
    g.A2[(n * 2 + j) * 16 + e] = s * sc;
  }
  // B1/B2: per x component
  for (int x = 0; x < 3; ++x) {
    {
      float sc = n3[0 * 2 + j]; sc *= sc;
      const float* Tb = g.T4 + (size_t)n * 1024 + (1 + x) * 64 + 0 * 32;
      const float* W = L3 + ((0 * 2 + j) * 32) * 16 + e;
      float s = 0.f;
      for (int c = 0; c < 32; ++c) s += Tb[c] * W[c * 16];
      g.B1[((n * 2 + j) * 3 + x) * 16 + e] = s * sc;
    }
    {
      float sc = n3[1 * 2 + j]; sc *= sc;
      const float* Tb = g.T4 + (size_t)n * 1024 + (1 + x) * 64 + 1 * 32;
      const float* W = L3 + ((1 * 2 + j) * 32) * 16 + e;
      float s = 0.f;
      for (int c = 0; c < 32; ++c) s += Tb[c] * W[c * 16];
      g.B2[((n * 2 + j) * 3 + x) * 16 + e] = s * sc;
    }
  }
}

// ========== kernel 5: F_nm assembly + embMP net (16->16->32->64) ===========
__global__ __launch_bounds__(256) void k_mp(MPArgs g) {
  __shared__ float bufA[8][16][32];
  __shared__ float bufB[8][16][32];
  const int w = threadIdx.x >> 5, lane = threadIdx.x & 31;
  const int half = lane >> 4, p = lane & 15, nn = lane & 15;
  const int n = blockIdx.y;
  const int m0 = blockIdx.x * 128 + w * 16;
  const int ti = (n >= HALFN) ? 1 : 0, tj = (m0 >= HALFN) ? 1 : 0;
  const int m = m0 + p;
  float (*BA)[32] = bufA[w];
  float (*BB)[32] = bufB[w];
  const MPW wM = g.Mw[ti][tj];

  float r3[3];
#pragma unroll
  for (int x = 0; x < 3; ++x) r3[x] = g.RX[(size_t)(1 + x) * NNp + n * Nn + m];
  const float* A1p = g.A1 + (n * 2 + tj) * 16;
  const float* A2p = g.A2 + (m * 2 + ti) * 16;
  const float* B1p = g.B1 + (n * 2 + tj) * 48;
  const float* B2p = g.B2 + (m * 2 + ti) * 48;

  v16h a;
#pragma unroll
  for (int k = 0; k < 8; ++k) {
    int c = half * 8 + k;
    float f = A1p[c] + A2p[c] + (float)g.embR[((size_t)n * Nn + m) * 16 + c];
#pragma unroll
    for (int x = 0; x < 3; ++x) f += r3[x] * (B1p[x * 16 + c] + B2p[x * 16 + c]);
    BA[p][c] = f;
    a[k] = (half_t)f;
  }
#pragma unroll
  for (int k = 8; k < 16; ++k) a[k] = (half_t)0.f;
  __syncthreads();

  // L0: 16 -> 16, h = x + tanh(xW+b)
  {
    v16h b;
#pragma unroll
    for (int e = 0; e < 16; ++e) {
      int K = kmapA(e, half);
      b[e] = (K < 16) ? (half_t)wM.W0[K * 16 + nn] : (half_t)0.f;
    }
    v8f acc = {};
    acc = wmma16(a, b, acc);
#pragma unroll
    for (int v = 0; v < 8; ++v) {
      int row = v + 8 * half;
      BB[row][nn] = BA[row][nn] + fast_tanh(acc[v] + wM.b0[nn]);
    }
  }
  __syncthreads();
  // L1: 16 -> 32, h = [x,x] + tanh(xW+b)
  {
    v16h a1;
#pragma unroll
    for (int e = 0; e < 16; ++e)
      a1[e] = (e < 8) ? (half_t)BB[p][(e & 7) + 8 * half] : (half_t)0.f;
#pragma unroll
    for (int cb = 0; cb < 2; ++cb) {
      v16h b;
#pragma unroll
      for (int e = 0; e < 16; ++e) {
        int K = kmapA(e, half);
        b[e] = (K < 16) ? (half_t)wM.W1[K * 32 + cb * 16 + nn] : (half_t)0.f;
      }
      v8f acc = {};
      acc = wmma16(a1, b, acc);
#pragma unroll
      for (int v = 0; v < 8; ++v) {
        int row = v + 8 * half;
        int ch = cb * 16 + nn;
        BA[row][ch] = BB[row][ch & 15] + fast_tanh(acc[v] + wM.b1[ch]);
      }
    }
  }
  __syncthreads();
  // L2: 32 -> 64, h = [x,x] + dt * tanh(xW+b)
  {
    v16h a2;
#pragma unroll
    for (int e = 0; e < 16; ++e) a2[e] = (half_t)BA[p][kmapA(e, half)];
#pragma unroll
    for (int cb = 0; cb < 4; ++cb) {
      v16h b;
#pragma unroll
      for (int e = 0; e < 16; ++e)
        b[e] = (half_t)wM.W2[kmapA(e, half) * 64 + cb * 16 + nn];
      v8f acc = {};
      acc = wmma16(a2, b, acc);
#pragma unroll
      for (int v = 0; v < 8; ++v) {
        int row = v + 8 * half;
        int ch = cb * 16 + nn;
        float val = BA[row][ch & 31] + wM.dt[ch] * fast_tanh(acc[v] + wM.b2[ch]);
        g.out[((size_t)n * Nn + m0 + row) * 64 + ch] = (half_t)val;
      }
    }
  }
}

// ============== kernel 7: descriptor G[n][12*64] from T_NXC ================
__global__ __launch_bounds__(256) void k_g(const float* __restrict__ TX,
                                           const float* __restrict__ Tbias,
                                           float* __restrict__ G) {
  int id = blockIdx.x * blockDim.x + threadIdx.x;
  if (id >= Nn * 64) return;
  int n = id >> 6, b = id & 63;
  const float* Tb = TX + (size_t)n * 1024;
  float tnc_b = Tb[b] + Tbias[b];
  float t3[3], t6[6];
#pragma unroll
  for (int x = 0; x < 3; ++x) t3[x] = Tb[(1 + x) * 64 + b];
#pragma unroll
  for (int x = 0; x < 6; ++x) t6[x] = Tb[(4 + x) * 64 + b];
  float* Gn = G + (size_t)n * 768;
#pragma unroll
  for (int a = 0; a < 4; ++a) Gn[a * 64 + b] = tnc_b * (Tb[a] + Tbias[a]);
#pragma unroll
  for (int a = 0; a < 4; ++a) {
    float s = 0.f;
#pragma unroll
    for (int x = 0; x < 3; ++x) s += t3[x] * Tb[(1 + x) * 64 + 4 + a];
    Gn[(4 + a) * 64 + b] = s;
  }
  float g1[3][2];
#pragma unroll
  for (int x = 0; x < 3; ++x)
#pragma unroll
    for (int j = 0; j < 2; ++j) g1[x][j] = Tb[(1 + x) * 64 + 8 + j];
  float g11[6][2];
#pragma unroll
  for (int x = 0; x < 3; ++x)
#pragma unroll
    for (int j = 0; j < 2; ++j) {
      g11[x][j] = g1[x][j] * g1[x][j];
      g11[3 + x][j] = 1.41421356237f * g1[x][j] * g1[(x + 1) % 3][j];
    }
#pragma unroll
  for (int j = 0; j < 2; ++j) {
    float s = 0.f;
#pragma unroll
    for (int xx = 0; xx < 6; ++xx) s += g11[xx][j] * t6[xx];
    Gn[(8 + j) * 64 + b] = s;
  }
#pragma unroll
  for (int j = 0; j < 2; ++j) {
    float s = 0.f;
#pragma unroll
    for (int xx = 0; xx < 6; ++xx) s += Tb[(4 + xx) * 64 + 10 + j] * t6[xx];
    Gn[(10 + j) * 64 + b] = s;
  }
}

// ============ kernel 8: fitting net 768 -> 240 -> 240(+x) -> 1 =============
// one wave per 16-atom tile; 15 C-tile accumulators (240 cols)
__global__ __launch_bounds__(32) void k_fit(FitArgs g) {
  __shared__ float h1s[16][256];
  __shared__ float h2s[16][256];
  const int lane = threadIdx.x & 31;
  const int half = lane >> 4, p = lane & 15, nn = lane & 15;
  const int t = blockIdx.x / 24, tile = blockIdx.x % 24;
  const int atom0 = t * HALFN + tile * 16;
  const FitW fw = g.Fw[t];
  // zero-pad columns 240..255
  for (int idx = lane; idx < 256; idx += 32) {
    h1s[idx >> 4][240 + (idx & 15)] = 0.f;
    h2s[idx >> 4][240 + (idx & 15)] = 0.f;
  }
  __syncthreads();

  // layer0: K=768
  v8f acc[15] = {};
  for (int kc = 0; kc < 24; ++kc) {
    if (kc + 1 < 24)
      __builtin_prefetch(&fw.W0[(size_t)((kc + 1) * 32 + lane) * 240], 0, 1);
    v16h a;
#pragma unroll
    for (int e = 0; e < 16; ++e)
      a[e] = (half_t)g.G[(size_t)(atom0 + p) * 768 + kc * 32 + kmapA(e, half)];
#pragma unroll
    for (int cb = 0; cb < 15; ++cb) {
      v16h b;
#pragma unroll
      for (int e = 0; e < 16; ++e)
        b[e] = (half_t)fw.W0[(size_t)(kc * 32 + kmapA(e, half)) * 240 + cb * 16 + nn];
      acc[cb] = wmma16(a, b, acc[cb]);
    }
  }
#pragma unroll
  for (int cb = 0; cb < 15; ++cb)
#pragma unroll
    for (int v = 0; v < 8; ++v) {
      int row = v + 8 * half, col = cb * 16 + nn;
      h1s[row][col] = fast_tanh(acc[cb][v] + fw.b0[col]);
    }
  __syncthreads();

  // layer1: K=240 (padded to 256), h2 = h1 + tanh(h1 W + b)
  v8f acc2[15] = {};
  for (int kc = 0; kc < 8; ++kc) {
    v16h a;
#pragma unroll
    for (int e = 0; e < 16; ++e)
      a[e] = (half_t)h1s[p][kc * 32 + kmapA(e, half)];
#pragma unroll
    for (int cb = 0; cb < 15; ++cb) {
      v16h b;
#pragma unroll
      for (int e = 0; e < 16; ++e) {
        int K = kc * 32 + kmapA(e, half);
        b[e] = (K < 240) ? (half_t)fw.W1[(size_t)K * 240 + cb * 16 + nn]
                         : (half_t)0.f;
      }
      acc2[cb] = wmma16(a, b, acc2[cb]);
    }
  }
#pragma unroll
  for (int cb = 0; cb < 15; ++cb)
#pragma unroll
    for (int v = 0; v < 8; ++v) {
      int row = v + 8 * half, col = cb * 16 + nn;
      h2s[row][col] = h1s[row][col] + fast_tanh(acc2[cb][v] + fw.b1[col]);
    }
  __syncthreads();

  // layer2: 240 -> 1 (linear) + Ebias
  if (lane < 16) {
    float s = fw.b2[0] + g.Eb[t];
    for (int col = 0; col < 240; ++col) s += h2s[lane][col] * fw.W2[col];
    g.atomE[atom0 + lane] = s;
  }
}

// ========================= kernel 9: final reduce ==========================
__global__ __launch_bounds__(256) void k_reduce(const float* __restrict__ atomE,
                                                float* __restrict__ out) {
  __shared__ float sbuf[256];
  float s = 0.f;
  for (int i = threadIdx.x; i < Nn; i += 256) s += atomE[i];
  sbuf[threadIdx.x] = s;
  __syncthreads();
  for (int st = 128; st > 0; st >>= 1) {
    if (threadIdx.x < st) sbuf[threadIdx.x] += sbuf[threadIdx.x + st];
    __syncthreads();
  }
  if (threadIdx.x == 0) out[0] = sbuf[0];
}

// ============================== host launcher ==============================
extern "C" void kernel_launch(void* const* d_in, const int* in_sizes, int n_in,
                              void* d_out, int out_size, void* d_ws, size_t ws_size,
                              hipStream_t stream) {
  int q = 0;
  auto nxt = [&]() { return (const float*)d_in[q++]; };

  // leaf order: coord, box, then params pytree in sorted-key order:
  // Ebias, Tbias, embA[i][j]{W0,W1,b0,b1}, embB..., embMP[i][j]{W0,W1,W2,b0,b1,b2,dt2},
  // embR[i][j]{W0,W1,W2,b0,b1,b2}, fit[t]{W0,W1,W2,b0,b1,b2}, lin1, lin3, n1, n3
  const float* coord = nxt();
  const float* box   = nxt();
  const float* Ebias = nxt();
  const float* Tbias = nxt();

  EmbedArgs ea;
  for (int i = 0; i < 2; ++i)
    for (int j = 0; j < 2; ++j) {
      ea.A[i][j].W0 = nxt(); ea.A[i][j].W1 = nxt();
      ea.A[i][j].b0 = nxt(); ea.A[i][j].b1 = nxt();
    }
  for (int i = 0; i < 2; ++i)
    for (int j = 0; j < 2; ++j) {
      ea.Bw[i][j].W0 = nxt(); ea.Bw[i][j].W1 = nxt();
      ea.Bw[i][j].b0 = nxt(); ea.Bw[i][j].b1 = nxt();
    }
  MPArgs mp;
  for (int i = 0; i < 2; ++i)
    for (int j = 0; j < 2; ++j) {
      mp.Mw[i][j].W0 = nxt(); mp.Mw[i][j].W1 = nxt(); mp.Mw[i][j].W2 = nxt();
      mp.Mw[i][j].b0 = nxt(); mp.Mw[i][j].b1 = nxt(); mp.Mw[i][j].b2 = nxt();
      mp.Mw[i][j].dt = nxt();
    }
  for (int i = 0; i < 2; ++i)
    for (int j = 0; j < 2; ++j) {
      ea.R[i][j].W0 = nxt(); ea.R[i][j].W1 = nxt(); ea.R[i][j].W2 = nxt();
      ea.R[i][j].b0 = nxt(); ea.R[i][j].b1 = nxt(); ea.R[i][j].b2 = nxt();
    }
  FitArgs fa;
  for (int t = 0; t < 2; ++t) {
    fa.Fw[t].W0 = nxt(); fa.Fw[t].W1 = nxt(); fa.Fw[t].W2 = nxt();
    fa.Fw[t].b0 = nxt(); fa.Fw[t].b1 = nxt(); fa.Fw[t].b2 = nxt();
  }
  ProjArgs pa;
  pa.lin1[0] = nxt(); pa.lin1[1] = nxt();
  pa.lin3[0] = nxt(); pa.lin3[1] = nxt();
  pa.n1v[0]  = nxt(); pa.n1v[1]  = nxt();
  pa.n3v[0]  = nxt(); pa.n3v[1]  = nxt();

  // ------------------------- workspace carve-up ---------------------------
  char* wsb = (char*)d_ws;
  size_t off = 0;
  auto alloc = [&](size_t bytes) -> void* {
    void* p = wsb + off;
    off = (off + bytes + 255) & ~(size_t)255;
    return p;
  };
  float*  s_cn  = (float*) alloc((size_t)NNp * 4);            //  2.4 MB
  float*  RX    = (float*) alloc((size_t)10 * NNp * 4);       // 23.6 MB
  half_t* embed = (half_t*)alloc((size_t)NNp * 64 * 2);       // 75.5 MB (reused for embMP)
  half_t* embRb = (half_t*)alloc((size_t)NNp * 16 * 2);       // 18.9 MB
  float*  T4    = (float*) alloc((size_t)Nn * 1024 * 4);      //  3.1 MB
  float*  TX    = (float*) alloc((size_t)Nn * 1024 * 4);      //  3.1 MB
  float*  D2    = (float*) alloc((size_t)Nn * 256 * 4);
  float*  A1    = (float*) alloc((size_t)Nn * 2 * 16 * 4);
  float*  A2    = (float*) alloc((size_t)Nn * 2 * 16 * 4);
  float*  B1    = (float*) alloc((size_t)Nn * 2 * 48 * 4);
  float*  B2    = (float*) alloc((size_t)Nn * 2 * 48 * 4);
  float*  G     = (float*) alloc((size_t)Nn * 768 * 4);       //  2.4 MB
  float*  atomE = (float*) alloc((size_t)Nn * 4);

  ea.s_cn = s_cn; ea.embed = embed; ea.embR = embRb;
  mp.RX = RX; mp.embR = embRb; mp.A1 = A1; mp.A2 = A2; mp.B1 = B1; mp.B2 = B2;
  mp.out = embed;  // embed consumed by first contract; reuse buffer for embMP
  pa.D2 = D2; pa.T4 = T4; pa.A1 = A1; pa.A2 = A2; pa.B1 = B1; pa.B2 = B2;
  fa.Eb = Ebias; fa.G = G; fa.atomE = atomE;

  // ------------------------------ pipeline --------------------------------
  k_geom<<<(NNp + 255) / 256, 256, 0, stream>>>(coord, box, s_cn, RX);
  k_embed<<<dim3(6, Nn), 256, 0, stream>>>(ea);
  k_contract<<<Nn / 8, 256, 0, stream>>>(RX, embed, T4, 4);
  k_d2<<<Nn, 256, 0, stream>>>(T4, D2);
  k_proj<<<(Nn * 32 + 255) / 256, 256, 0, stream>>>(pa);
  k_mp<<<dim3(6, Nn), 256, 0, stream>>>(mp);
  k_contract<<<Nn / 8, 256, 0, stream>>>(RX, embed, TX, 10);
  k_g<<<(Nn * 64 + 255) / 256, 256, 0, stream>>>(TX, Tbias, G);
  k_fit<<<48, 32, 0, stream>>>(fa);
  k_reduce<<<1, 256, 0, stream>>>(atomE, (float*)d_out);
}
